// GraphNeuralNetwork_39633958207658
// MI455X (gfx1250) — matrix operand order repaired and checked by reading, not compiled, not run on hardware
//
#include <hip/hip_runtime.h>

typedef __attribute__((ext_vector_type(16))) _Float16 v16h;
typedef __attribute__((ext_vector_type(8)))  _Float16 v8h;
typedef __attribute__((ext_vector_type(8)))  float    v8f;
typedef __attribute__((ext_vector_type(4)))  float    v4f;

#define N_NODES   100000
#define N_EDGES   800000
#define DIM       128
#define N_CLASSES 4
#define LDS_STRIDE 136   // 128 + 8 halves pad -> conflict-free strided LDS reads

// ---------------------------------------------------------------------------
// Convert fp32 weights to f16 once per launch (32KB each, stays in L0/L2)
// ---------------------------------------------------------------------------
__global__ void convert_w_kernel(const float* __restrict__ w0,
                                 const float* __restrict__ w1,
                                 _Float16* __restrict__ w0h,
                                 _Float16* __restrict__ w1h) {
    int i = blockIdx.x * blockDim.x + threadIdx.x;
    if (i < DIM * DIM) {
        w0h[i] = (_Float16)w0[i];
        w1h[i] = (_Float16)w1[i];
    }
}

// ---------------------------------------------------------------------------
// Fused GEMM + bias + ReLU:  Out[n,j] = relu( sum_k A[n,k]*Wh[j,k] + bias[j] )
// One wave per 16x16 output tile, 8 waves/block cover 16 rows x 128 cols.
// A tile staged to LDS as f16; B fragments read directly from f16 weights.
// ---------------------------------------------------------------------------
__global__ __launch_bounds__(256)
void gemm_relu_kernel(const float* __restrict__ A,      // [N, 128] fp32
                      const _Float16* __restrict__ Wh,  // [128 out, 128 in] f16
                      const float* __restrict__ bias,   // [128]
                      float* __restrict__ Out) {        // [N, 128]
    __shared__ _Float16 As[16 * LDS_STRIDE];

    const int tid     = threadIdx.x;
    const int wave    = tid >> 5;        // 0..7  -> output-column tile
    const int lane    = tid & 31;
    const int rowBase = blockIdx.x * 16;

    // Stage 16x128 activation tile into LDS (fp32 -> f16), coalesced.
    for (int i = tid; i < 16 * DIM; i += 256) {
        int r = i >> 7, c = i & 127;
        As[r * LDS_STRIDE + c] = (_Float16)A[(rowBase + r) * DIM + c];
    }
    __syncthreads();

    const int m     = lane & 15;   // A row within tile == B/C column within tile
    const int hi    = lane >> 4;   // half-wave selector per ISA fragment layout
    const int jbase = wave * 16;   // global output-feature base for this wave
    const _Float16* wrow = Wh + (jbase + m) * DIM;

    v8f acc = {};
    #pragma unroll
    for (int k0 = 0; k0 < DIM; k0 += 32) {
        // A fragment: lane holds row m, 16 contiguous K at k0+16*hi
        v8h a0 = *(const v8h*)(&As[m * LDS_STRIDE + k0 + 16 * hi]);
        v8h a1 = *(const v8h*)(&As[m * LDS_STRIDE + k0 + 16 * hi + 8]);
        v16h a;
        #pragma unroll
        for (int i = 0; i < 8; ++i) { a[i] = a0[i]; a[i + 8] = a1[i]; }
        // B fragment: lane holds column m, 16 contiguous K at k0+16*hi
        v16h b = *(const v16h*)(wrow + k0 + 16 * hi);
        acc = __builtin_amdgcn_wmma_f32_16x16x32_f16(
            /*neg_a=*/false, a, /*neg_b=*/false, b,
            /*c_mod=*/(short)0, acc, /*reuse_a=*/false, /*reuse_b=*/false);
    }

    // Epilogue: acc[v] = C[M = v + 8*hi][N = m]
    const float bv = bias[jbase + m];
    #pragma unroll
    for (int v = 0; v < 8; ++v) {
        int   mrow = v + 8 * hi;
        float val  = acc[v] + bv;
        val = val > 0.0f ? val : 0.0f;
        Out[(rowBase + mrow) * DIM + jbase + m] = val;
    }
}

// ---------------------------------------------------------------------------
// Degree: deg[row[e]] += 1
// ---------------------------------------------------------------------------
__global__ void degree_kernel(const int* __restrict__ rows,
                              float* __restrict__ deg) {
    int e = blockIdx.x * blockDim.x + threadIdx.x;
    if (e < N_EDGES) atomicAdd(&deg[rows[e]], 1.0f);
}

// ---------------------------------------------------------------------------
// Scatter-add: agg[row[e], f] += H[col[e], f]
// One thread per (edge, feature); each 128-thread group streams one
// contiguous 512B source row and hits contiguous atomic destinations (L2).
// ---------------------------------------------------------------------------
__global__ __launch_bounds__(256)
void scatter_kernel(const int* __restrict__ rows,
                    const int* __restrict__ cols,
                    const float* __restrict__ H,
                    float* __restrict__ agg) {
    unsigned gid = blockIdx.x * 256u + threadIdx.x;   // < 102,400,000
    int e = (int)(gid >> 7);
    int f = (int)(gid & 127u);
    if (e < N_EDGES) {
        atomicAdd(&agg[rows[e] * DIM + f], H[cols[e] * DIM + f]);
    }
}

// ---------------------------------------------------------------------------
// Residual + degree normalization, in place into agg:
//   h[n,f] = H[n,f] + agg[n,f] / max(deg[n], 1)
// ---------------------------------------------------------------------------
__global__ void combine_kernel(const float* __restrict__ H,
                               const float* __restrict__ deg,
                               float* __restrict__ agg) {
    int i = blockIdx.x * blockDim.x + threadIdx.x;
    if (i < N_NODES * DIM) {
        float d = deg[i >> 7];
        d = d < 1.0f ? 1.0f : d;
        agg[i] = H[i] + agg[i] / d;
    }
}

// ---------------------------------------------------------------------------
// Classifier: logits[n,c] = sum_k H[n,k]*Wc[c,k] + bc[c]
// One wave per node: coalesced float4 row load + shfl_xor reduction.
// ---------------------------------------------------------------------------
__global__ __launch_bounds__(256)
void classifier_kernel(const float* __restrict__ H,
                       const float* __restrict__ Wc,
                       const float* __restrict__ bc,
                       float* __restrict__ out) {
    int node = (int)((blockIdx.x * 256u + threadIdx.x) >> 5);
    int lane = threadIdx.x & 31;
    if (node >= N_NODES) return;

    v4f h = *(const v4f*)(H + node * DIM + lane * 4);
    float p0 = 0.f, p1 = 0.f, p2 = 0.f, p3 = 0.f;
    #pragma unroll
    for (int i = 0; i < 4; ++i) {
        float hv = h[i];
        int   k  = lane * 4 + i;
        p0 += hv * Wc[0 * DIM + k];
        p1 += hv * Wc[1 * DIM + k];
        p2 += hv * Wc[2 * DIM + k];
        p3 += hv * Wc[3 * DIM + k];
    }
    #pragma unroll
    for (int off = 16; off; off >>= 1) {
        p0 += __shfl_xor(p0, off, 32);
        p1 += __shfl_xor(p1, off, 32);
        p2 += __shfl_xor(p2, off, 32);
        p3 += __shfl_xor(p3, off, 32);
    }
    if (lane == 0) {
        out[node * 4 + 0] = p0 + bc[0];
        out[node * 4 + 1] = p1 + bc[1];
        out[node * 4 + 2] = p2 + bc[2];
        out[node * 4 + 3] = p3 + bc[3];
    }
}

// ---------------------------------------------------------------------------
extern "C" void kernel_launch(void* const* d_in, const int* in_sizes, int n_in,
                              void* d_out, int out_size, void* d_ws, size_t ws_size,
                              hipStream_t stream) {
    const float* x  = (const float*)d_in[0];
    const int*   ei = (const int*)  d_in[1];   // [2, N_EDGES]: rows then cols
    const float* W0 = (const float*)d_in[2];
    const float* b0 = (const float*)d_in[3];
    const float* W1 = (const float*)d_in[4];
    const float* b1 = (const float*)d_in[5];
    const float* Wc = (const float*)d_in[6];
    const float* bc = (const float*)d_in[7];
    float* out = (float*)d_out;

    const int* rows = ei;
    const int* cols = ei + N_EDGES;

    // Workspace carve (all offsets 256B-aligned)
    char*  ws         = (char*)d_ws;
    size_t feat_bytes = (size_t)N_NODES * DIM * sizeof(float);          // 51.2 MB
    size_t deg_bytes  = (((size_t)N_NODES * sizeof(float)) + 255) & ~(size_t)255;
    float*    h0   = (float*)(ws);
    float*    hbuf = (float*)(ws + feat_bytes);                          // agg -> h
    float*    deg  = (float*)(ws + 2 * feat_bytes);
    _Float16* w0h  = (_Float16*)(ws + 2 * feat_bytes + deg_bytes);
    _Float16* w1h  = w0h + DIM * DIM;

    // Zero accumulators (capturable memset nodes)
    hipMemsetAsync(hbuf, 0, feat_bytes, stream);
    hipMemsetAsync(deg,  0, (size_t)N_NODES * sizeof(float), stream);

    // f16 weight copies
    convert_w_kernel<<<(DIM * DIM + 255) / 256, 256, 0, stream>>>(W0, W1, w0h, w1h);

    // Layer 0: h0 = relu(x @ W0^T + b0)
    gemm_relu_kernel<<<N_NODES / 16, 256, 0, stream>>>(x, w0h, b0, h0);

    // Degree + scatter aggregation
    degree_kernel<<<(N_EDGES + 255) / 256, 256, 0, stream>>>(rows, deg);
    scatter_kernel<<<(N_EDGES * (DIM / 256)) * 2, 256, 0, stream>>>(rows, cols, h0, hbuf);

    // Residual + normalize: hbuf = h0 + hbuf/deg
    combine_kernel<<<(N_NODES * DIM + 255) / 256, 256, 0, stream>>>(h0, deg, hbuf);

    // Layer 1: h0 = relu(hbuf @ W1^T + b1)   (reuse h0 buffer)
    gemm_relu_kernel<<<N_NODES / 16, 256, 0, stream>>>(hbuf, w1h, b1, h0);

    // Classifier: out = h0 @ Wc^T + bc
    classifier_kernel<<<(N_NODES * 32 + 255) / 256, 256, 0, stream>>>(h0, Wc, bc, out);
}
